// CausalSelfAttention_85366769975916
// MI455X (gfx1250) — matrix operand order repaired and checked by reading, not compiled, not run on hardware
//
#include <hip/hip_runtime.h>
#include <hip/hip_bf16.h>

// Problem constants (match reference)
#define BATCH 4
#define SEQ   2048
#define EMB   1024
#define NHEAD 16
#define HDIM  64
#define EMB3  3072
#define ROWS  (BATCH * SEQ)   // 8192

typedef __attribute__((ext_vector_type(16))) __bf16 v16bf;
typedef __attribute__((ext_vector_type(8)))  __bf16 v8bf;
typedef __attribute__((ext_vector_type(4)))  __bf16 v4bf;
typedef __attribute__((ext_vector_type(8)))  float  v8f;
typedef __attribute__((ext_vector_type(4)))  float  v4f;
typedef __attribute__((ext_vector_type(4)))  unsigned int v4u;
typedef __attribute__((ext_vector_type(8)))  int    v8i;
typedef __attribute__((ext_vector_type(4)))  int    v4i;

#define WMMA_BF16(A_, B_, C_) \
  __builtin_amdgcn_wmma_f32_16x16x32_bf16(false, (A_), false, (B_), (short)0, (C_), false, false)

#define TDM_AVAILABLE __has_builtin(__builtin_amdgcn_tensor_load_to_lds)

__device__ __forceinline__ v16bf make16(v8bf lo, v8bf hi) {
  v16bf r;
#pragma unroll
  for (int i = 0; i < 8; ++i) { r[i] = lo[i]; r[i + 8] = hi[i]; }
  return r;
}

// A-fragment (16x32 bf16, row = lane&15): lane holds k = half*16 + e, contiguous
__device__ __forceinline__ v16bf load_a32(const __bf16* base, int ld, int m0, int k0, int lane) {
  int m = m0 + (lane & 15), hf = lane >> 4;
  const __bf16* p = base + (size_t)m * ld + k0 + hf * 16;
  return make16(*(const v8bf*)p, *(const v8bf*)(p + 8));
}

// B-fragment (32x16 bf16, col = lane&15): lane holds k = half*16 + e, contiguous from B^T row
__device__ __forceinline__ v16bf load_b32T(const __bf16* baseT, int ld, int n0, int k0, int lane) {
  int n = n0 + (lane & 15), hf = lane >> 4;
  const __bf16* p = baseT + (size_t)n * ld + k0 + hf * 16;
  return make16(*(const v8bf*)p, *(const v8bf*)(p + 8));
}

// generic pointer to an LDS object: low 32 bits == LDS byte offset (aperture layout)
__device__ __forceinline__ unsigned ldsAddrOf(const void* p) {
  return (unsigned)(size_t)p;
}

// Issue a 2D TDM tile load: tileD0 (contiguous elems) x tileD1 (rows), bf16 elements,
// row stride = strideElems, tensor bounds tensorD0 x tensorD1 (for OOB). D# per ISA ch.8.
__device__ __forceinline__ void tdm_load_2d(unsigned ldsOff, const void* gTileStart,
                                            unsigned tileD0, unsigned tileD1,
                                            unsigned long long strideElems,
                                            unsigned tensorD0, unsigned tensorD1) {
#if TDM_AVAILABLE
  unsigned long long ga = (unsigned long long)(size_t)gTileStart;
  v4u g0;
  g0[0] = 1u;                                                  // count=1 valid descriptor
  g0[1] = ldsOff;                                              // lds_addr [63:32]
  g0[2] = (unsigned)(ga & 0xFFFFFFFFu);                        // global_addr [95:64]
  g0[3] = (unsigned)((ga >> 32) & 0x01FFFFFFu) | (2u << 30);   // addr hi + type=2 [127:126]
  v8i g1;
  g1[0] = (int)(1u << 16);                                     // data_size=1 -> 2 bytes
  g1[1] = (int)((tensorD0 & 0xFFFFu) << 16);                   // tensor_dim0 [79:48] lo
  g1[2] = (int)(((tensorD0 >> 16) & 0xFFFFu) |                 // tensor_dim0 hi
                ((tensorD1 & 0xFFFFu) << 16));                 // tensor_dim1 lo
  g1[3] = (int)(((tensorD1 >> 16) & 0xFFFFu) |                 // tensor_dim1 hi
                ((tileD0 & 0xFFFFu) << 16));                   // tile_dim0 [127:112]
  g1[4] = (int)(tileD1 & 0xFFFFu);                             // tile_dim1, tile_dim2=0
  g1[5] = (int)(strideElems & 0xFFFFFFFFull);                  // dim0 stride lo
  g1[6] = (int)((strideElems >> 32) & 0xFFFFull);              // dim0 stride hi
  g1[7] = 0;
  v4i gz = {0, 0, 0, 0};
#if defined(__clang_major__) && (__clang_major__ >= 23)
  v8i gz8 = {};
  __builtin_amdgcn_tensor_load_to_lds(g0, g1, gz, gz, gz8, 0);
#else
  __builtin_amdgcn_tensor_load_to_lds(g0, g1, gz, gz, 0);
#endif
#endif
}

// ---------------- conversion kernels ----------------

__global__ __launch_bounds__(256) void cvt_f32_to_bf16_4(const float* __restrict__ in,
                                                         __bf16* __restrict__ out, int n4) {
  int i = blockIdx.x * blockDim.x + threadIdx.x;
  if (i >= n4) return;
  v4f v = ((const v4f*)in)[i];
  v4bf o;
#pragma unroll
  for (int j = 0; j < 4; ++j) o[j] = (__bf16)v[j];
  ((v4bf*)out)[i] = o;
}

// out[n*K + k] = (bf16) in[k*N + n]   (weight [K,N] -> transposed bf16 [N,K])
__global__ __launch_bounds__(256) void cvt_transpose_bf16(const float* __restrict__ in,
                                                          __bf16* __restrict__ out, int K, int N) {
  int idx = blockIdx.x * blockDim.x + threadIdx.x;
  if (idx >= K * N) return;
  int k = idx % K;
  int n = idx / K;
  out[(size_t)n * K + k] = (__bf16)in[(size_t)k * N + n];
}

// vT[b][h][d][t] = qkv[b*T+t][2C + h*64 + d]
__global__ __launch_bounds__(256) void transpose_v_bf16(const __bf16* __restrict__ qkv,
                                                        __bf16* __restrict__ vt) {
  int idx = blockIdx.x * blockDim.x + threadIdx.x;
  if (idx >= ROWS * EMB) return;
  int t = idx % SEQ;
  int d = (idx / SEQ) % HDIM;
  int h = (idx / (SEQ * HDIM)) % NHEAD;
  int b = idx / (SEQ * HDIM * NHEAD);
  vt[idx] = qkv[(size_t)(b * SEQ + t) * EMB3 + 2 * EMB + h * HDIM + d];
}

// ---------------- block-cooperative WMMA GEMM ----------------
// C[MxN] = A[MxK] * B, with B given transposed [N,K].
// Block = 8 waves -> 128(M) x 64(N) tile. The 64x32 bf16 B-panel per k-step is
// staged in double-buffered LDS via TDM (tensor_load_to_lds), overlapped with
// WMMA compute; A-fragments are software-pipelined from global.

__global__ __launch_bounds__(256) void gemm_bf16_lds(const __bf16* __restrict__ A,
                                                     const __bf16* __restrict__ Bt,
                                                     __bf16* __restrict__ Cbf,
                                                     float* __restrict__ Cf,
                                                     int M, int N, int K) {
  __shared__ __align__(16) __bf16 bTile[2][64 * 32];  // 8 KB double buffer
  int wv = threadIdx.x >> 5, lane = threadIdx.x & 31;
  int nGroups = N >> 6;
  int bm = blockIdx.x / nGroups, ng = blockIdx.x % nGroups;
  int m0 = bm * 128 + wv * 16;
  int n0 = ng << 6;
  (void)M;

  // preload k-step 0 panel into buffer 0
#if TDM_AVAILABLE
  if (threadIdx.x < 32) {
    tdm_load_2d(ldsAddrOf(&bTile[0][0]), Bt + (size_t)n0 * K, 32, 64,
                (unsigned long long)K, (unsigned)K, (unsigned)N);
    __builtin_amdgcn_s_wait_tensorcnt(0);
  }
#else
  {
    int r = threadIdx.x >> 2, c = (threadIdx.x & 3) * 8;
    *(v8bf*)(&bTile[0][r * 32 + c]) = *(const v8bf*)(Bt + (size_t)(n0 + r) * K + c);
  }
#endif
  __syncthreads();

  v8f acc[4] = {};
  v16bf a_cur = load_a32(A, K, m0, 0, lane);
  int nk = K >> 5;
  for (int ki = 0; ki < nk; ++ki) {
    int cur = ki & 1, nxt = cur ^ 1;
    int k0 = ki << 5;
    // kick off next B panel DMA (completes during this k-step's compute)
    if (ki + 1 < nk) {
#if TDM_AVAILABLE
      if (threadIdx.x < 32)
        tdm_load_2d(ldsAddrOf(&bTile[nxt][0]), Bt + (size_t)n0 * K + (k0 + 32), 32, 64,
                    (unsigned long long)K, (unsigned)K, (unsigned)N);
#else
      int r = threadIdx.x >> 2, c = (threadIdx.x & 3) * 8;
      *(v8bf*)(&bTile[nxt][r * 32 + c]) =
          *(const v8bf*)(Bt + (size_t)(n0 + r) * K + (k0 + 32) + c);
#endif
    }
    // software-pipelined A fragment for the next k-step
    int kn = (ki + 1 < nk) ? (k0 + 32) : k0;
    v16bf a_nxt = load_a32(A, K, m0, kn, lane);

    // 4 WMMA from the LDS-resident B panel
#pragma unroll
    for (int t = 0; t < 4; ++t) {
      const __bf16* p = &bTile[cur][(t * 16 + (lane & 15)) * 32 + (lane >> 4) * 16];
      v16bf b = make16(*(const v8bf*)p, *(const v8bf*)(p + 8));
      acc[t] = WMMA_BF16(a_cur, b, acc[t]);
    }
    a_cur = a_nxt;
#if TDM_AVAILABLE
    if (threadIdx.x < 32) __builtin_amdgcn_s_wait_tensorcnt(0);
#endif
    __syncthreads();
  }

  int n = lane & 15, hf = lane >> 4;
#pragma unroll
  for (int j = 0; j < 8; ++j) {
    size_t row = (size_t)(m0 + j + 8 * hf) * N;
    if (Cf) {
#pragma unroll
      for (int t = 0; t < 4; ++t) Cf[row + n0 + t * 16 + n] = acc[t][j];
    } else {
#pragma unroll
      for (int t = 0; t < 4; ++t) Cbf[row + n0 + t * 16 + n] = (__bf16)acc[t][j];
    }
  }
}

// ---------------- flash attention: one wave per (b, h, 16-query tile) ----------------

__global__ __launch_bounds__(256) void flash_attn_bf16(const __bf16* __restrict__ qkv,
                                                       const __bf16* __restrict__ vt,
                                                       __bf16* __restrict__ y) {
  __shared__ __align__(16) __bf16 plds[8][16][32];  // per-wave P staging (C-layout -> A-layout)
  int w = threadIdx.x >> 5;
  int lane = threadIdx.x & 31;
  int wid = blockIdx.x * 8 + w;
  int qt = wid & (SEQ / 16 - 1);          // 128 query tiles
  int h  = (wid >> 7) & (NHEAD - 1);
  int b  = wid >> 11;
  int q0 = qt << 4;

  const __bf16* qb  = qkv + (size_t)b * SEQ * EMB3 + (size_t)h * HDIM;
  const __bf16* kbp = qkv + (size_t)b * SEQ * EMB3 + EMB + (size_t)h * HDIM;
  const __bf16* vtb = vt + (size_t)((b * NHEAD + h) * HDIM) * SEQ;  // [64][T]

  int n = lane & 15, hf = lane >> 4;

  // Q fragments for d = 0..31 and 32..63 (A layout, row = query)
  v16bf qa0, qa1;
  {
    const __bf16* qr = qb + (size_t)(q0 + n) * EMB3;
    const __bf16* p0 = qr + hf * 16;
    qa0 = make16(*(const v8bf*)p0, *(const v8bf*)(p0 + 8));
    const __bf16* p1 = qr + 32 + hf * 16;
    qa1 = make16(*(const v8bf*)p1, *(const v8bf*)(p1 + 8));
  }

  v8f o0 = {}, o1 = {}, o2 = {}, o3 = {};  // 16x64 output accumulator
  float rm[8], rs[8];
#pragma unroll
  for (int j = 0; j < 8; ++j) { rm[j] = -1e30f; rs[j] = 0.0f; }
  const float scale = 0.125f;  // 1/sqrt(64)

  for (int kb = 0; kb < q0 + 16; kb += 32) {  // kb <= q0 always (causal coverage)
    // S = Q * K^T for 32 keys -> two 16x16 tiles
    v8f s1 = {}, s2 = {};
    {
      const __bf16* kr1 = kbp + (size_t)(kb + n) * EMB3;        // keys kb..kb+15
      const __bf16* kr2 = kbp + (size_t)(kb + 16 + n) * EMB3;   // keys kb+16..kb+31
      const __bf16* p;
      v16bf bk;
      p = kr1 + hf * 16;       bk = make16(*(const v8bf*)p, *(const v8bf*)(p + 8)); s1 = WMMA_BF16(qa0, bk, s1);
      p = kr1 + 32 + hf * 16;  bk = make16(*(const v8bf*)p, *(const v8bf*)(p + 8)); s1 = WMMA_BF16(qa1, bk, s1);
      p = kr2 + hf * 16;       bk = make16(*(const v8bf*)p, *(const v8bf*)(p + 8)); s2 = WMMA_BF16(qa0, bk, s2);
      p = kr2 + 32 + hf * 16;  bk = make16(*(const v8bf*)p, *(const v8bf*)(p + 8)); s2 = WMMA_BF16(qa1, bk, s2);
    }

    // causal mask + online softmax (rows live across the 16 lanes of each half)
    float corr[8];
#pragma unroll
    for (int j = 0; j < 8; ++j) {
      int q = q0 + j + 8 * hf;
      float e1 = (kb + n <= q)      ? s1[j] * scale : -1e30f;
      float e2 = (kb + 16 + n <= q) ? s2[j] * scale : -1e30f;
      float mx = fmaxf(e1, e2);
      mx = fmaxf(mx, __shfl_xor(mx, 1, 32));
      mx = fmaxf(mx, __shfl_xor(mx, 2, 32));
      mx = fmaxf(mx, __shfl_xor(mx, 4, 32));
      mx = fmaxf(mx, __shfl_xor(mx, 8, 32));
      float nm = fmaxf(rm[j], mx);
      float c  = __expf(rm[j] - nm);
      float p1 = __expf(e1 - nm);
      float p2 = __expf(e2 - nm);
      float ps = p1 + p2;
      ps += __shfl_xor(ps, 1, 32);
      ps += __shfl_xor(ps, 2, 32);
      ps += __shfl_xor(ps, 4, 32);
      ps += __shfl_xor(ps, 8, 32);
      rs[j] = rs[j] * c + ps;
      rm[j] = nm;
      corr[j] = c;
      s1[j] = p1;
      s2[j] = p2;
    }
#pragma unroll
    for (int j = 0; j < 8; ++j) {
      o0[j] *= corr[j]; o1[j] *= corr[j]; o2[j] *= corr[j]; o3[j] *= corr[j];
    }

    // P (C layout) -> LDS -> reload as A fragment. DS ops are in-order within a wave.
#pragma unroll
    for (int j = 0; j < 8; ++j) {
      plds[w][j + 8 * hf][n]      = (__bf16)s1[j];
      plds[w][j + 8 * hf][16 + n] = (__bf16)s2[j];
    }
    v16bf pa;
    {
      const __bf16* pr = &plds[w][n][0];  // row = query (lane&15)
      const __bf16* p0 = pr + hf * 16;
      pa = make16(*(const v8bf*)p0, *(const v8bf*)(p0 + 8));
    }

    // O += P * V ; B fragment from vT (k = key dim, contiguous)
#pragma unroll
    for (int t = 0; t < 4; ++t) {
      const __bf16* p = vtb + (size_t)(t * 16 + n) * SEQ + kb + hf * 16;
      v16bf bv = make16(*(const v8bf*)p, *(const v8bf*)(p + 8));
      if (t == 0) o0 = WMMA_BF16(pa, bv, o0);
      if (t == 1) o1 = WMMA_BF16(pa, bv, o1);
      if (t == 2) o2 = WMMA_BF16(pa, bv, o2);
      if (t == 3) o3 = WMMA_BF16(pa, bv, o3);
    }
  }

  // normalize + store y (bf16, [rows, C], head h at columns h*64..)
#pragma unroll
  for (int j = 0; j < 8; ++j) {
    size_t row = (size_t)(b * SEQ + q0 + j + 8 * hf) * EMB;
    float inv = 1.0f / rs[j];
    y[row + h * HDIM + 0 * 16 + n] = (__bf16)(o0[j] * inv);
    y[row + h * HDIM + 1 * 16 + n] = (__bf16)(o1[j] * inv);
    y[row + h * HDIM + 2 * 16 + n] = (__bf16)(o2[j] * inv);
    y[row + h * HDIM + 3 * 16 + n] = (__bf16)(o3[j] * inv);
  }
}

// ---------------- launch ----------------

extern "C" void kernel_launch(void* const* d_in, const int* in_sizes, int n_in,
                              void* d_out, int out_size, void* d_ws, size_t ws_size,
                              hipStream_t stream) {
  (void)in_sizes; (void)n_in; (void)out_size; (void)ws_size;
  const float* x      = (const float*)d_in[0];
  const float* w_attn = (const float*)d_in[1];
  const float* w_proj = (const float*)d_in[2];
  float* out = (float*)d_out;

  char* ws = (char*)d_ws;
  // workspace layout (bytes)
  __bf16* xb   = (__bf16*)(ws);                 // 16 MB  : x bf16 [8192,1024]
  __bf16* waT  = (__bf16*)(ws + 16777216);      //  6 MB  : w_attn^T bf16 [3072,1024]
  __bf16* wpT  = (__bf16*)(ws + 23068672);      //  2 MB  : w_proj^T bf16 [1024,1024]
  __bf16* qkvb = (__bf16*)(ws + 25165824);      // 48 MB  : qkv bf16 [8192,3072]
  __bf16* vtb  = (__bf16*)(ws + 75497472);      // 16 MB  : V^T bf16 [4,16,64,2048]
  __bf16* yb   = (__bf16*)(ws + 92274688);      // 16 MB  : attn out bf16 [8192,1024]

  // 1. precision conversion / weight transpose
  {
    int n4 = ROWS * EMB / 4;
    cvt_f32_to_bf16_4<<<(n4 + 255) / 256, 256, 0, stream>>>(x, xb, n4);
  }
  {
    int tot = EMB * EMB3;
    cvt_transpose_bf16<<<(tot + 255) / 256, 256, 0, stream>>>(w_attn, waT, EMB, EMB3);
  }
  {
    int tot = EMB * EMB;
    cvt_transpose_bf16<<<(tot + 255) / 256, 256, 0, stream>>>(w_proj, wpT, EMB, EMB);
  }
  // 2. QKV GEMM: [8192,3072] = [8192,1024] x [1024,3072]; block = 128x64 tile
  {
    int blocks = (ROWS / 128) * (EMB3 / 64);  // 64 * 48 = 3072
    gemm_bf16_lds<<<blocks, 256, 0, stream>>>(xb, waT, qkvb, nullptr, ROWS, EMB3, EMB);
  }
  // 3. V transpose for P*V B-fragments
  {
    int tot = ROWS * EMB;
    transpose_v_bf16<<<(tot + 255) / 256, 256, 0, stream>>>(qkvb, vtb);
  }
  // 4. flash attention
  {
    int waves = BATCH * NHEAD * (SEQ / 16);  // 8192
    flash_attn_bf16<<<waves / 8, 256, 0, stream>>>(qkvb, vtb, yb);
  }
  // 5. output projection: [8192,1024] f32 = [8192,1024] x [1024,1024]
  {
    int blocks = (ROWS / 128) * (EMB / 64);  // 64 * 16 = 1024
    gemm_bf16_lds<<<blocks, 256, 0, stream>>>(yb, wpT, nullptr, out, ROWS, EMB, EMB);
  }
}
